// MultiHeadAttention_31628139168211
// MI455X (gfx1250) — matrix operand order repaired and checked by reading, not compiled, not run on hardware
//
#include <hip/hip_runtime.h>
#include <hip/hip_bf16.h>

// MI455X / gfx1250 multi-head causal attention (B=2, S=2048, D=1024, H=16, dh=64)
// ~95 GFLOP vs ~60 MB of traffic (fits in 192MB L2, 23.3 TB/s HBM) => purely
// matrix-op bound. Everything runs on v_wmma_f32_16x16x32_f16 (f16 in, f32 acc).

typedef __attribute__((ext_vector_type(16))) _Float16 v16h;
typedef __attribute__((ext_vector_type(8)))  _Float16 v8h;
typedef __attribute__((ext_vector_type(8)))  float    v8f;
typedef __attribute__((ext_vector_type(4)))  float    v4f;

constexpr int BB = 2;     // batch
constexpr int SS = 2048;  // sequence
constexpr int DD = 1024;  // model dim
constexpr int HH = 16;    // heads
constexpr int DH = 64;    // head dim

__device__ __forceinline__ v16h join8(v8h lo, v8h hi) {
  v16h r;
  #pragma unroll
  for (int i = 0; i < 8; ++i) { r[i] = lo[i]; r[8 + i] = hi[i]; }
  return r;
}

__device__ __forceinline__ float halfmax(float v) {
  #pragma unroll
  for (int off = 1; off < 16; off <<= 1) v = fmaxf(v, __shfl_xor(v, off, 32));
  return v;  // reduces within each 16-lane half of the wave32
}
__device__ __forceinline__ float halfsum(float v) {
  #pragma unroll
  for (int off = 1; off < 16; off <<= 1) v += __shfl_xor(v, off, 32);
  return v;
}

// ---------------------------------------------------------------------------
// Kernel 1: convert Wq/Wk/Wv (f32, [k][n]) -> WT (f16, [mat][n][k]) so the
// GEMM B-operand (lane = N column, K contiguous) loads are coalesced v8h.
// ---------------------------------------------------------------------------
__global__ void wt_transpose_kernel(const float* __restrict__ Wq,
                                    const float* __restrict__ Wk,
                                    const float* __restrict__ Wv,
                                    _Float16* __restrict__ wt) {
  size_t idx = (size_t)blockIdx.x * blockDim.x + threadIdx.x;
  const size_t per = (size_t)DD * DD;
  if (idx >= 3 * per) return;
  int mat = (int)(idx / per);
  size_t rem = idx - (size_t)mat * per;
  int n = (int)(rem / DD);
  int k = (int)(rem % DD);
  const float* w = (mat == 0) ? Wq : (mat == 1) ? Wk : Wv;
  wt[idx] = (_Float16)w[(size_t)k * DD + n];
}

// ---------------------------------------------------------------------------
// Kernel 2: QKV projection. One wave computes a 16x64 tile of x@W: per 32-wide
// K step, ONE A-tile (16 f32->f16 cvts) feeds FOUR v_wmma_f32_16x16x32_f16.
// Q,K stored f16 [B,H,S,dh]; V stored f16 [B,H,dh,S] (transposed) so the
// attention P@V B-operand is contiguous.
// ---------------------------------------------------------------------------
__global__ __launch_bounds__(32)
void qkv_proj_kernel(const float* __restrict__ x,
                     const _Float16* __restrict__ wt,
                     _Float16* __restrict__ qo,
                     _Float16* __restrict__ ko,
                     _Float16* __restrict__ vo) {
  const int lane = threadIdx.x;
  const int col  = lane & 15;   // A row index / B,N column / C N column
  const int hi   = lane >> 4;   // half-wave select
  const int n0   = blockIdx.x * 64;      // output column tile in [0,D)
  const int gr0  = blockIdx.y * 16;      // global row tile in [0,B*S)
  const int mat  = blockIdx.z;           // 0=Q 1=K 2=V

  const float*    xr = x + (size_t)(gr0 + col) * DD;  // A: row = lane&15
  const _Float16* wb = wt + (size_t)mat * DD * DD;

  v8f c[4];
  #pragma unroll
  for (int nn = 0; nn < 4; ++nn) c[nn] = {};

  for (int k0 = 0; k0 < DD; k0 += 32) {
    // A operand (16x32 f16): lanes<16 hold K {0..7,16..23}, lanes>=16 {8..15,24..31}
    const v4f* pa0 = (const v4f*)(xr + k0 + hi * 8);
    const v4f* pa1 = (const v4f*)(xr + k0 + 16 + hi * 8);
    v4f f0 = pa0[0], f1 = pa0[1], f2 = pa1[0], f3 = pa1[1];
    v16h a;
    #pragma unroll
    for (int i = 0; i < 4; ++i) {
      a[i]      = (_Float16)f0[i];
      a[4 + i]  = (_Float16)f1[i];
      a[8 + i]  = (_Float16)f2[i];
      a[12 + i] = (_Float16)f3[i];
    }
    // B operand (32x16 f16): lane n holds column n; lanes<16 K {0..7,16..23}, lanes>=16 {8..15,24..31}
    #pragma unroll
    for (int nn = 0; nn < 4; ++nn) {
      const v8h* pb = (const v8h*)(wb + (size_t)(n0 + nn * 16 + col) * DD + k0 + hi * 8);
      v16h bm = join8(pb[0], pb[2]);
      c[nn] = __builtin_amdgcn_wmma_f32_16x16x32_f16(false, a, false, bm,
                                                     (short)0, c[nn], false, false);
    }
  }

  // C layout: VGPR r, lanes<16 -> M=r, lanes>=16 -> M=r+8, N=lane&15
  // Hoisted destination select (no per-store branching).
  if (mat != 2) {
    _Float16* dst = (mat == 0) ? qo : ko;
    #pragma unroll
    for (int nn = 0; nn < 4; ++nn) {
      const int ncol = n0 + nn * 16 + col;
      const int h = ncol >> 6, d = ncol & 63;
      #pragma unroll
      for (int r = 0; r < 8; ++r) {
        const int gr = gr0 + r + 8 * hi;
        const int b = gr >> 11, s = gr & 2047;
        dst[((size_t)(b * HH + h) * SS + s) * DH + d] = (_Float16)c[nn][r];
      }
    }
  } else {
    #pragma unroll
    for (int nn = 0; nn < 4; ++nn) {
      const int ncol = n0 + nn * 16 + col;
      const int h = ncol >> 6, d = ncol & 63;
      #pragma unroll
      for (int r = 0; r < 8; ++r) {
        const int gr = gr0 + r + 8 * hi;
        const int b = gr >> 11, s = gr & 2047;
        vo[((size_t)(b * HH + h) * DH + d) * SS + s] = (_Float16)c[nn][r];  // transposed
      }
    }
  }
}

// ---------------------------------------------------------------------------
// Kernel 3: causal flash attention. One wave per (b, h, 16-query tile).
// 32-key blocks: 4 WMMAs for Q@K^T, online softmax, P transposed C->A layout
// through LDS, 4 WMMAs for P@V into 4 resident accumulator tiles.
// ---------------------------------------------------------------------------
__global__ __launch_bounds__(32)
void flash_attn_kernel(const _Float16* __restrict__ Q,
                       const _Float16* __restrict__ K,
                       const _Float16* __restrict__ VT,
                       float* __restrict__ out) {
  const int lane = threadIdx.x;
  const int col  = lane & 15;
  const int hi   = lane >> 4;
  const int q0   = blockIdx.x * 16;
  const int h    = blockIdx.y;
  const int b    = blockIdx.z;

  const _Float16* qp = Q  + (size_t)(b * HH + h) * SS * DH;
  const _Float16* kp = K  + (size_t)(b * HH + h) * SS * DH;
  const _Float16* vp = VT + (size_t)(b * HH + h) * DH * SS;  // [dh][S]

  __shared__ alignas(32) _Float16 plds[16 * 32];  // one wave per block

  // Resident Q A-tiles for dh chunks [0,32) and [32,64)
  v16h aq[2];
  #pragma unroll
  for (int cc = 0; cc < 2; ++cc) {
    const _Float16* qr = qp + (size_t)(q0 + col) * DH + cc * 32;
    aq[cc] = join8(*(const v8h*)(qr + hi * 8), *(const v8h*)(qr + 16 + hi * 8));
  }

  float mrow[8], lrow[8];
  v8f acc[4];
  #pragma unroll
  for (int r = 0; r < 8; ++r) { mrow[r] = -INFINITY; lrow[r] = 0.0f; }
  #pragma unroll
  for (int j = 0; j < 4; ++j) acc[j] = {};

  const float scale = 0.125f;  // 1/sqrt(64)

  for (int kb = 0; kb <= q0 + 15; kb += 32) {
    if (kb + 32 <= q0 + 15) {   // speculative prefetch of next K/V block
      __builtin_prefetch(kp + (size_t)(kb + 32 + col) * DH, 0, 0);
      __builtin_prefetch(vp + (size_t)col * SS + kb + 32, 0, 0);
    }

    // ---- scores: S = Q @ K^T, two 16-key N-tiles -------------------------
    v8f sc[2];
    #pragma unroll
    for (int t = 0; t < 2; ++t) {
      v8f c0 = {};
      #pragma unroll
      for (int cc = 0; cc < 2; ++cc) {
        // B operand: lane n = key column; lanes<16 dims cc*32+{0..7,16..23}, lanes>=16 +{8..15,24..31}
        const _Float16* kr = kp + (size_t)(kb + t * 16 + col) * DH + cc * 32 + hi * 8;
        v16h bm = join8(((const v8h*)kr)[0], ((const v8h*)kr)[2]);
        c0 = __builtin_amdgcn_wmma_f32_16x16x32_f16(false, aq[cc], false, bm,
                                                    (short)0, c0, false, false);
      }
      sc[t] = c0;
    }

    // ---- causal mask + scale --------------------------------------------
    #pragma unroll
    for (int t = 0; t < 2; ++t) {
      const int key = kb + t * 16 + col;
      #pragma unroll
      for (int r = 0; r < 8; ++r) {
        const int qr = q0 + r + 8 * hi;
        sc[t][r] = (key <= qr) ? sc[t][r] * scale : -INFINITY;
      }
    }

    // ---- online softmax: row max, alpha ---------------------------------
    float alpha[8];
    #pragma unroll
    for (int r = 0; r < 8; ++r) {
      float v  = halfmax(fmaxf(sc[0][r], sc[1][r]));
      float mn = fmaxf(mrow[r], v);
      alpha[r] = __expf(mrow[r] - mn);   // exp(-inf)=0 on first block
      mrow[r]  = mn;
    }

    __syncthreads();  // previous-iter LDS reads done before overwrite

    // ---- p = exp(s - m), store C-layout -> LDS, row sums ----------------
    #pragma unroll
    for (int r = 0; r < 8; ++r) {
      const float p0 = __expf(sc[0][r] - mrow[r]);
      const float p1 = __expf(sc[1][r] - mrow[r]);
      plds[(r + 8 * hi) * 32 + col]      = (_Float16)p0;
      plds[(r + 8 * hi) * 32 + 16 + col] = (_Float16)p1;
      lrow[r] = lrow[r] * alpha[r] + halfsum(p0 + p1);
    }
    #pragma unroll
    for (int j = 0; j < 4; ++j)
      #pragma unroll
      for (int r = 0; r < 8; ++r) acc[j][r] *= alpha[r];

    __syncthreads();  // LDS stores visible before A-layout reload

    // ---- reload P in A layout (16x32): row = lane&15 --------------------
    v16h pa = join8(*(const v8h*)&plds[col * 32 + hi * 8],
                    *(const v8h*)&plds[col * 32 + 16 + hi * 8]);

    // ---- acc += P @ V ; V^T layout makes B-operand contiguous -----------
    #pragma unroll
    for (int j = 0; j < 4; ++j) {
      const _Float16* vr = vp + (size_t)(16 * j + col) * SS + kb + hi * 8;
      v16h bm = join8(((const v8h*)vr)[0], ((const v8h*)vr)[2]);
      acc[j] = __builtin_amdgcn_wmma_f32_16x16x32_f16(false, pa, false, bm,
                                                      (short)0, acc[j], false, false);
    }
  }

  // ---- normalize and write out[b][s][h*64 + d] (f32) ----------------------
  float inv[8];
  #pragma unroll
  for (int r = 0; r < 8; ++r) inv[r] = 1.0f / lrow[r];
  #pragma unroll
  for (int j = 0; j < 4; ++j)
    #pragma unroll
    for (int r = 0; r < 8; ++r) {
      const int s = q0 + r + 8 * hi;
      out[((size_t)b * SS + s) * DD + h * DH + j * 16 + col] = acc[j][r] * inv[r];
    }
}

// ---------------------------------------------------------------------------
extern "C" void kernel_launch(void* const* d_in, const int* in_sizes, int n_in,
                              void* d_out, int out_size, void* d_ws, size_t ws_size,
                              hipStream_t stream) {
  const float* x  = (const float*)d_in[0];
  const float* Wq = (const float*)d_in[1];
  const float* Wk = (const float*)d_in[2];
  const float* Wv = (const float*)d_in[3];
  float* out = (float*)d_out;

  // workspace layout (f16): WT[3][D][D] | Q[B,H,S,dh] | K[B,H,S,dh] | V^T[B,H,dh,S]
  char* ws = (char*)d_ws;
  const size_t wtBytes  = 3ull * DD * DD * sizeof(_Float16);      // 6 MB
  const size_t qkvElems = (size_t)BB * HH * SS * DH;              // 4M elems
  _Float16* wt = (_Float16*)ws;
  _Float16* q  = (_Float16*)(ws + wtBytes);
  _Float16* k  = q + qkvElems;
  _Float16* v  = k + qkvElems;

  // 1) transpose+convert weights
  {
    const size_t n = 3ull * DD * DD;
    wt_transpose_kernel<<<dim3((unsigned)((n + 255) / 256)), dim3(256), 0, stream>>>(
        Wq, Wk, Wv, wt);
  }
  // 2) QKV projection (wave per 16x64 tile: 4 WMMAs per A-tile)
  qkv_proj_kernel<<<dim3(DD / 64, (BB * SS) / 16, 3), dim3(32), 0, stream>>>(
      x, wt, q, k, v);
  // 3) causal flash attention (wave per 16-query tile)
  flash_attn_kernel<<<dim3(SS / 16, HH, BB), dim3(32), 0, stream>>>(
      q, k, v, out);
}
